// MyDecoderLayer_54666343743584
// MI455X (gfx1250) — matrix operand order repaired and checked by reading, not compiled, not run on hardware
//
#include <hip/hip_runtime.h>
#include <stdint.h>
#include <stddef.h>

// ---------------------------------------------------------------------------
// MI455X (gfx1250, wave32) implementation of the MyDecoderLayer reference.
// All big GEMMs run on v_wmma_f32_16x16x32_bf16 (fp32 -> bf16 conversion in
// LDS staging, f32 accumulate). Elementwise/reduction work is fused.
// GEMM uses ping-pong LDS buffers: tile k+1 is fetched into registers while
// tile k's WMMAs run; one workgroup barrier per K-step.
// ---------------------------------------------------------------------------

typedef __bf16 v16bf __attribute__((ext_vector_type(16)));
typedef __bf16 v8bf  __attribute__((ext_vector_type(8)));
typedef __bf16 v2bf  __attribute__((ext_vector_type(2)));
typedef float  v8f   __attribute__((ext_vector_type(8)));

#define NT    4096      // tokens per batch
#define DIM   256       // D
#define DIM2  512       // 2D
#define C2D   1024      // 4D MixFFN hidden
#define HCN   8         // heads
#define BSZ   4         // batch
#define LPAD  40        // LDS row pitch in bf16 (80B = 20 banks; 16B aligned)

// RNE fp32 pair -> packed bf16x2 in one v_cvt_pk_bf16_f32 where available
__device__ __forceinline__ uint32_t pack2bf(float lo, float hi) {
#if __has_builtin(__builtin_amdgcn_cvt_pk_bf16_f32)
  auto p = __builtin_amdgcn_cvt_pk_bf16_f32(lo, hi);
  return __builtin_bit_cast(uint32_t, p);
#else
  v2bf p = { (__bf16)lo, (__bf16)hi };
  union { v2bf v; uint32_t u; } c; c.v = p; return c.u;
#endif
}

// ---------------------------------------------------------------------------
// Generic WMMA GEMM: C[b] = op(A[b]) @ W (+bias) (+R[b])
//   A: fp32, row-major [M,K] (transA=0, ldA=K) or stored [K,M] (transA=1,
//      ldA = M-stride) so A[m,k] = Ab[k*ldA + m]  (used for 'bdn,de->bne').
//   W: fp32 row-major [K,N] (shared across batch).
// Block: 256 threads = 8 waves. Block tile 128x64. Wave tile 32x32 (2x2 WMMA).
// Requires M%128==0, N%64==0, K%32==0 (true for all call sites here).
// LDS tiles are stored so each lane's WMMA fragment halves are contiguous
// 16-byte runs -> ds_load_b128 per half (A row-major, B transposed).
// ---------------------------------------------------------------------------
__global__ __launch_bounds__(256)
void gemm_bf16_wmma_kernel(const float* __restrict__ A, long long batchStrideA,
                           int transA, int ldA,
                           const float* __restrict__ W,
                           const float* __restrict__ bias,
                           const float* __restrict__ R,
                           float* __restrict__ C, long long batchStrideC,
                           int N, int K)
{
  __shared__ __attribute__((aligned(16))) __bf16 sA[2][128][LPAD];   // [m][k]
  __shared__ __attribute__((aligned(16))) __bf16 sBt[2][64][LPAD];   // [n][k]

  const int t    = threadIdx.x;
  const int wid  = t >> 5;
  const int lane = t & 31;
  const int hlf  = lane >> 4;      // lane half (wave32)
  const int r    = lane & 15;

  const int m0 = blockIdx.y * 128;
  const int n0 = blockIdx.x * 64;

  const float* Ab = A + (size_t)blockIdx.z * (size_t)batchStrideA;
  float*       Cb = C + (size_t)blockIdx.z * (size_t)batchStrideC;
  const float* Rb = R ? (R + (size_t)blockIdx.z * (size_t)batchStrideC) : nullptr;

  const int waveM = (wid >> 1) << 5;  // 0,32,64,96
  const int waveN = (wid & 1) << 5;   // 0,32

  const int rA0 = waveM + r;
  const int rA1 = waveM + 16 + r;
  const int cB0 = waveN + r;
  const int cB1 = waveN + 16 + r;

  // Per-thread staging assignments (loop-invariant).
  // A (no trans): thread covers row aRow, 16 consecutive k starting at aK0.
  const int aRow = t >> 1;
  const int aK0  = (t & 1) << 4;
  // A (trans): thread covers column k = tK, 16 consecutive m starting at tM0.
  const int tK   = t >> 3;
  const int tM0  = (t & 7) << 4;
  // B: thread covers output column bN, 8 consecutive k starting at bK0.
  const int bN   = t & 63;
  const int bK0  = (t >> 6) << 3;

  const float* aPtrN = Ab + (size_t)(m0 + aRow) * ldA + aK0;        // + kb
  const float* aPtrT = Ab + m0 + tM0;                               // + (kb+tK)*ldA
  const float* bPtr  = W + (size_t)bK0 * N + (n0 + bN);             // + kb*N

  float aq[16];   // staged A values (next tile)
  float bq[8];    // staged B values (next tile)

  auto load_tile = [&](int kb) {
    if (!transA) {
      const float4* s4 = (const float4*)(aPtrN + kb);
      float4 q0 = s4[0], q1 = s4[1], q2 = s4[2], q3 = s4[3];
      aq[0]=q0.x; aq[1]=q0.y; aq[2]=q0.z; aq[3]=q0.w;
      aq[4]=q1.x; aq[5]=q1.y; aq[6]=q1.z; aq[7]=q1.w;
      aq[8]=q2.x; aq[9]=q2.y; aq[10]=q2.z; aq[11]=q2.w;
      aq[12]=q3.x; aq[13]=q3.y; aq[14]=q3.z; aq[15]=q3.w;
    } else {
      const float4* s4 = (const float4*)(aPtrT + (size_t)(kb + tK) * ldA);
      float4 q0 = s4[0], q1 = s4[1], q2 = s4[2], q3 = s4[3];
      aq[0]=q0.x; aq[1]=q0.y; aq[2]=q0.z; aq[3]=q0.w;
      aq[4]=q1.x; aq[5]=q1.y; aq[6]=q1.z; aq[7]=q1.w;
      aq[8]=q2.x; aq[9]=q2.y; aq[10]=q2.z; aq[11]=q2.w;
      aq[12]=q3.x; aq[13]=q3.y; aq[14]=q3.z; aq[15]=q3.w;
    }
    const float* s = bPtr + (size_t)kb * N;
    #pragma unroll
    for (int j = 0; j < 8; ++j) bq[j] = s[(size_t)j * N];
  };

  auto store_tile = [&](int buf) {
    if (!transA) {
      uint32_t* dst = (uint32_t*)&sA[buf][aRow][aK0];
      #pragma unroll
      for (int j = 0; j < 8; ++j) dst[j] = pack2bf(aq[2 * j], aq[2 * j + 1]);
    } else {
      #pragma unroll
      for (int j = 0; j < 16; ++j) sA[buf][tM0 + j][tK] = (__bf16)aq[j];
    }
    uint32_t* dst = (uint32_t*)&sBt[buf][bN][bK0];
    #pragma unroll
    for (int j = 0; j < 4; ++j) dst[j] = pack2bf(bq[2 * j], bq[2 * j + 1]);
  };

  v8f acc00 = {0.f,0.f,0.f,0.f,0.f,0.f,0.f,0.f};
  v8f acc01 = acc00, acc10 = acc00, acc11 = acc00;

  // prologue: stage tile 0
  load_tile(0);
  store_tile(0);
  __syncthreads();

  for (int kb = 0; kb < K; kb += 32) {
    const int cur = (kb >> 5) & 1;
    const bool have_next = (kb + 32) < K;

    // issue global loads for the next tile; latency overlaps the WMMAs below
    if (have_next) load_tile(kb + 32);

    // ---- build fragments: two 16B contiguous runs per operand (b128 loads) --
    // A 16x32 bf16 layout: lane half 0 -> K 0..7 | 16..23 ; half 1 -> 8..15 | 24..31
    v8bf a0lo = *(const v8bf*)&sA[cur][rA0][hlf << 3];
    v8bf a0hi = *(const v8bf*)&sA[cur][rA0][16 + (hlf << 3)];
    v8bf a1lo = *(const v8bf*)&sA[cur][rA1][hlf << 3];
    v8bf a1hi = *(const v8bf*)&sA[cur][rA1][16 + (hlf << 3)];
    // B 32x16 bf16 layout: lane half 0 -> K 0..7 | 16..23 ; half 1 -> 8..15 | 24..31
    v8bf b0lo = *(const v8bf*)&sBt[cur][cB0][hlf << 3];
    v8bf b0hi = *(const v8bf*)&sBt[cur][cB0][16 + (hlf << 3)];
    v8bf b1lo = *(const v8bf*)&sBt[cur][cB1][hlf << 3];
    v8bf b1hi = *(const v8bf*)&sBt[cur][cB1][16 + (hlf << 3)];

    v16bf a0 = __builtin_shufflevector(a0lo, a0hi, 0,1,2,3,4,5,6,7,8,9,10,11,12,13,14,15);
    v16bf a1 = __builtin_shufflevector(a1lo, a1hi, 0,1,2,3,4,5,6,7,8,9,10,11,12,13,14,15);
    v16bf b0 = __builtin_shufflevector(b0lo, b0hi, 0,1,2,3,4,5,6,7,8,9,10,11,12,13,14,15);
    v16bf b1 = __builtin_shufflevector(b1lo, b1hi, 0,1,2,3,4,5,6,7,8,9,10,11,12,13,14,15);

    acc00 = __builtin_amdgcn_wmma_f32_16x16x32_bf16(false, a0, false, b0, (short)0, acc00, false, false);
    acc01 = __builtin_amdgcn_wmma_f32_16x16x32_bf16(false, a0, false, b1, (short)0, acc01, false, false);
    acc10 = __builtin_amdgcn_wmma_f32_16x16x32_bf16(false, a1, false, b0, (short)0, acc10, false, false);
    acc11 = __builtin_amdgcn_wmma_f32_16x16x32_bf16(false, a1, false, b1, (short)0, acc11, false, false);

    // convert+store the next tile into the other LDS buffer (no RAW vs 'cur')
    if (have_next) store_tile(cur ^ 1);
    __syncthreads();
  }

  // ---- epilogue: C layout (8 VGPRs): lane<16 -> M=v, lane>=16 -> M=v+8 ----
  #pragma unroll
  for (int mi = 0; mi < 2; ++mi) {
    int rowbase = m0 + waveM + mi * 16 + (hlf << 3);
    #pragma unroll
    for (int ni = 0; ni < 2; ++ni) {
      int col = n0 + waveN + ni * 16 + r;
      v8f acc = (mi == 0) ? ((ni == 0) ? acc00 : acc01)
                          : ((ni == 0) ? acc10 : acc11);
      float bv = bias ? bias[col] : 0.0f;
      #pragma unroll
      for (int v = 0; v < 8; ++v) {
        size_t idx = (size_t)(rowbase + v) * N + col;
        float val = acc[v] + bv;
        if (Rb) val += Rb[idx];
        Cb[idx] = val;
      }
    }
  }
}

// ---------------------------------------------------------------------------
// Row LayerNorm (optionally followed by exact GELU). One block per row.
// cols <= 1024; values register-cached so in-place use is safe.
// ---------------------------------------------------------------------------
__global__ __launch_bounds__(256)
void layernorm_kernel(const float* __restrict__ src,
                      const float* __restrict__ gam,
                      const float* __restrict__ bet,
                      float* __restrict__ dst, int cols, int doGelu)
{
  __shared__ float rs[256];
  __shared__ float rq[256];
  const int t = threadIdx.x;
  const size_t row = blockIdx.x;
  const float* s = src + row * (size_t)cols;
  float* d = dst + row * (size_t)cols;
  float vals[4];
  float sum = 0.f, sq = 0.f;
  const int iters = (cols + 255) >> 8;
  for (int i = 0; i < iters; ++i) {
    int c = i * 256 + t;
    float v = (c < cols) ? s[c] : 0.f;
    vals[i] = v;
    sum += v; sq += v * v;
  }
  rs[t] = sum; rq[t] = sq;
  __syncthreads();
  for (int st = 128; st > 0; st >>= 1) {
    if (t < st) { rs[t] += rs[t + st]; rq[t] += rq[t + st]; }
    __syncthreads();
  }
  const float inv  = 1.0f / (float)cols;
  const float mean = rs[0] * inv;
  const float var  = rq[0] * inv - mean * mean;
  const float rstd = rsqrtf(var + 1e-5f);
  for (int i = 0; i < iters; ++i) {
    int c = i * 256 + t;
    if (c < cols) {
      float y = (vals[i] - mean) * rstd * gam[c] + bet[c];
      if (doGelu) y = 0.5f * y * (1.0f + erff(y * 0.70710678118654752f));
      d[c] = y;
    }
  }
}

// softmax over tokens (axis N) for one (b,d) channel column of n2[B,N,D];
// writes Ksm in [B, D, N] layout.
__global__ __launch_bounds__(256)
void softmax_tokens_kernel(const float* __restrict__ n2, float* __restrict__ Ksm)
{
  __shared__ float red[256];
  const int t  = threadIdx.x;
  const int bd = blockIdx.x;            // b*DIM + d
  const int b  = bd >> 8;
  const int d  = bd & 255;
  const float* s = n2 + (size_t)b * NT * DIM + d;    // stride DIM over tokens
  float* o = Ksm + (size_t)bd * NT;
  float mx = -3.4e38f;
  for (int n = t; n < NT; n += 256) mx = fmaxf(mx, s[(size_t)n * DIM]);
  red[t] = mx; __syncthreads();
  for (int st = 128; st > 0; st >>= 1) { if (t < st) red[t] = fmaxf(red[t], red[t + st]); __syncthreads(); }
  mx = red[0];
  __syncthreads();
  float sum = 0.f;
  for (int n = t; n < NT; n += 256) { float e = __expf(s[(size_t)n * DIM] - mx); o[n] = e; sum += e; }
  red[t] = sum; __syncthreads();
  for (int st = 128; st > 0; st >>= 1) { if (t < st) red[t] += red[t + st]; __syncthreads(); }
  const float inv = 1.0f / red[0];
  for (int n = t; n < NT; n += 256) o[n] *= inv;
}

// softmax over head channels (hk=32) per (b,h,token); writes Qsm in [B,D,N].
__global__ __launch_bounds__(256)
void softmax_channels_kernel(const float* __restrict__ n2, float* __restrict__ Qsm)
{
  const int idx = blockIdx.x * 256 + threadIdx.x;   // B*HC*NT
  const int n  = idx & 4095;
  const int h2 = (idx >> 12) & 7;
  const int b  = idx >> 15;
  const float* s = n2 + ((size_t)b * NT + n) * DIM + h2 * 32;
  float e[32];
  float mx = -3.4e38f;
  #pragma unroll
  for (int c = 0; c < 32; ++c) mx = fmaxf(mx, s[c]);
  float sum = 0.f;
  #pragma unroll
  for (int c = 0; c < 32; ++c) { e[c] = __expf(s[c] - mx); sum += e[c]; }
  const float inv = 1.0f / sum;
  float* o = Qsm + ((size_t)b * DIM + h2 * 32) * NT + n;
  #pragma unroll
  for (int c = 0; c < 32; ++c) o[(size_t)c * NT] = e[c] * inv;
}

// ctx[b,h,k,v] = sum_n Ksm[b,h*32+k,n] * n1[b,n,h*32+v]   (32x32 per head)
__global__ __launch_bounds__(256)
void ctx_kernel(const float* __restrict__ Ksm, const float* __restrict__ n1,
                float* __restrict__ ctx)
{
  const int bh = blockIdx.x;            // b*HC + h
  const int b  = bh >> 3, h2 = bh & 7;
  const int t  = threadIdx.x;
  const int k  = t >> 3;
  const int v0 = (t & 7) << 2;
  const float* kp = Ksm + ((size_t)b * DIM + h2 * 32 + k) * NT;
  const float* vp = n1 + (size_t)b * NT * DIM + h2 * 32 + v0;
  float a0 = 0.f, a1 = 0.f, a2 = 0.f, a3 = 0.f;
  for (int n = 0; n < NT; ++n) {
    float kv = kp[n];
    const float* vr = vp + (size_t)n * DIM;
    a0 = fmaf(kv, vr[0], a0);
    a1 = fmaf(kv, vr[1], a1);
    a2 = fmaf(kv, vr[2], a2);
    a3 = fmaf(kv, vr[3], a3);
  }
  float* op = ctx + (size_t)bh * 1024 + k * 32 + v0;
  op[0] = a0; op[1] = a1; op[2] = a2; op[3] = a3;
}

// att[b,h*32+v,n] = sum_k ctx[b,h,k,v] * Qsm[b,h*32+k,n]; writes [B,D,N].
__global__ __launch_bounds__(256)
void att_kernel(const float* __restrict__ ctx, const float* __restrict__ Qsm,
                float* __restrict__ att)
{
  __shared__ float sc[1024];
  const int bh = blockIdx.y;
  const int b  = bh >> 3, h2 = bh & 7;
  const int n  = blockIdx.x * 256 + threadIdx.x;
  for (int i = threadIdx.x; i < 1024; i += 256) sc[i] = ctx[(size_t)bh * 1024 + i];
  __syncthreads();
  float q[32];
  const float* qp = Qsm + ((size_t)b * DIM + h2 * 32) * NT + n;
  #pragma unroll
  for (int k = 0; k < 32; ++k) q[k] = qp[(size_t)k * NT];
  float* op = att + ((size_t)b * DIM + h2 * 32) * NT + n;
  #pragma unroll 4
  for (int v = 0; v < 32; ++v) {
    float a = 0.f;
    #pragma unroll
    for (int k = 0; k < 32; ++k) a = fmaf(sc[k * 32 + v], q[k], a);
    op[(size_t)v * NT] = a;
  }
}

// tx = concat(x1e, x2) + rep   (in place on rep buffer)
__global__ __launch_bounds__(256)
void tx_concat_add_kernel(const float* __restrict__ x1e, const float* __restrict__ x2,
                          float* __restrict__ tx)
{
  const size_t idx = (size_t)blockIdx.x * 256 + threadIdx.x;  // B*N*D2
  const int e = (int)(idx & 511);
  const size_t bn = idx >> 9;
  const float add = (e < DIM) ? x1e[bn * DIM + e] : x2[bn * DIM + (e - DIM)];
  tx[idx] += add;
}

// Depthwise 3x3x3 conv (SAME, zero pad) on [B,C2,16,16,16] fused with
// bias and the "+h" skip: out = conv(h) + b_dw + h. Data layout [B,N,C2].
__global__ __launch_bounds__(256)
void dwconv_kernel(const float* __restrict__ h, const float* __restrict__ w,
                   const float* __restrict__ bdw, float* __restrict__ out)
{
  const size_t idx = (size_t)blockIdx.x * 256 + threadIdx.x;  // B*N*C2
  const int c = (int)(idx & 1023);
  const size_t bn = idx >> 10;
  const int n = (int)(bn & 4095);
  const int b = (int)(bn >> 12);
  const int z = n >> 8, y = (n >> 4) & 15, x = n & 15;
  float acc = bdw[c] + h[idx];
  const float* wp = w + (size_t)c * 27;
  const float* hb = h + (size_t)b * NT * C2D + c;
  #pragma unroll
  for (int dz = -1; dz <= 1; ++dz) {
    const int zz = z + dz; if (zz < 0 || zz > 15) continue;
    #pragma unroll
    for (int dy = -1; dy <= 1; ++dy) {
      const int yy = y + dy; if (yy < 0 || yy > 15) continue;
      #pragma unroll
      for (int dx = -1; dx <= 1; ++dx) {
        const int xx = x + dx; if (xx < 0 || xx > 15) continue;
        const int nn = (zz << 8) | (yy << 4) | xx;
        acc = fmaf(wp[(dz + 1) * 9 + (dy + 1) * 3 + (dx + 1)],
                   hb[(size_t)nn * C2D], acc);
      }
    }
  }
  out[idx] = acc;
}

// ---------------------------------------------------------------------------
static void launch_gemm(const float* A, long long sA, int transA, int ldA,
                        const float* W, const float* bias, const float* R,
                        float* C, long long sC, int M, int N, int K, int batch,
                        hipStream_t stream)
{
  dim3 grid(N / 64, M / 128, batch), block(256);
  hipLaunchKernelGGL(gemm_bf16_wmma_kernel, grid, block, 0, stream,
                     A, sA, transA, ldA, W, bias, R, C, sC, N, K);
}

extern "C" void kernel_launch(void* const* d_in, const int* in_sizes, int n_in,
                              void* d_out, int out_size, void* d_ws, size_t ws_size,
                              hipStream_t stream)
{
  (void)in_sizes; (void)n_in; (void)out_size;

  const float* x1    = (const float*)d_in[0];
  const float* x2    = (const float*)d_in[1];
  const float* w_x1  = (const float*)d_in[2];
  const float* b_x1  = (const float*)d_in[3];
  const float* g_n1  = (const float*)d_in[4];
  const float* be_n1 = (const float*)d_in[5];
  const float* w_rep = (const float*)d_in[6];
  const float* b_rep = (const float*)d_in[7];
  const float* g_an  = (const float*)d_in[8];
  const float* be_an = (const float*)d_in[9];
  const float* g_n2  = (const float*)d_in[10];
  const float* be_n2 = (const float*)d_in[11];
  const float* w_fc1 = (const float*)d_in[12];
  const float* b_fc1 = (const float*)d_in[13];
  const float* w_dw  = (const float*)d_in[14];
  const float* b_dw  = (const float*)d_in[15];
  const float* g_m1  = (const float*)d_in[16];
  const float* be_m1 = (const float*)d_in[17];
  const float* w_fc2 = (const float*)d_in[18];
  const float* b_fc2 = (const float*)d_in[19];
  const float* w_cat = (const float*)d_in[20];
  const float* b_cat = (const float*)d_in[21];
  const float* w_exp = (const float*)d_in[22];
  const float* g_up  = (const float*)d_in[23];
  const float* be_up = (const float*)d_in[24];

  float* ws = (float*)d_ws;
  const size_t SZ = (size_t)BSZ * NT * DIM;             // 4,194,304 floats
  const size_t off_x1e  = 0;                            // x1e              (4M)
  const size_t off_n1   = off_x1e  + SZ;                // n1   -> att      (4M)
  const size_t off_n2   = off_n1   + SZ;                // n2   -> y        (4M)
  const size_t off_Ksm  = off_n2   + SZ;                // Ksm  -> ntx/mx   (4M)
  const size_t off_Qsm  = off_Ksm  + SZ;                // Qsm  (ntx/mx hi) (4M)
  const size_t off_ctx  = off_Qsm  + SZ;                // ctx              (64K)
  const size_t off_tx   = off_ctx  + 65536;             // rep -> tx        (8M)
  const size_t off_h    = off_tx   + 2 * SZ;            // h                (16M)
  const size_t off_dwph = off_h    + 4 * SZ;            // dw+h -> ye       (16M)
  const size_t off_ax   = off_dwph + 4 * SZ;            // ax               (16M)
  const size_t need = (off_ax + 4 * SZ) * sizeof(float);   // ~304 MB
  if (ws_size < need) return;   // workspace too small; cannot run

  const int MROWS = BSZ * NT;   // 16384

  // 1) x1e = x1 @ w_x1 + b_x1
  launch_gemm(x1, 0, 0, DIM, w_x1, b_x1, nullptr, ws + off_x1e, 0,
              MROWS, DIM, DIM, 1, stream);
  // 2) n1 = LN(x1e), n2 = LN(x2)  (shared norm1 params)
  hipLaunchKernelGGL(layernorm_kernel, dim3(MROWS), dim3(256), 0, stream,
                     ws + off_x1e, g_n1, be_n1, ws + off_n1, DIM, 0);
  hipLaunchKernelGGL(layernorm_kernel, dim3(MROWS), dim3(256), 0, stream,
                     x2, g_n1, be_n1, ws + off_n2, DIM, 0);
  // 3) K = softmax over tokens, Q = softmax over head channels
  hipLaunchKernelGGL(softmax_tokens_kernel, dim3(BSZ * DIM), dim3(256), 0, stream,
                     ws + off_n2, ws + off_Ksm);
  hipLaunchKernelGGL(softmax_channels_kernel, dim3((BSZ * HCN * NT) / 256), dim3(256), 0, stream,
                     ws + off_n2, ws + off_Qsm);
  // 4) ctx = K @ V^T per head; att = ctx^T @ Q  (att written over n1 region)
  hipLaunchKernelGGL(ctx_kernel, dim3(BSZ * HCN), dim3(256), 0, stream,
                     ws + off_Ksm, ws + off_n1, ws + off_ctx);
  hipLaunchKernelGGL(att_kernel, dim3(NT / 256, BSZ * HCN), dim3(256), 0, stream,
                     ws + off_ctx, ws + off_Qsm, ws + off_n1);
  // 5) rep = att^T @ w_rep + b_rep  ('bdn,de->bne'), per-batch transA GEMM
  launch_gemm(ws + off_n1, (long long)DIM * NT, 1, NT,
              w_rep, b_rep, nullptr, ws + off_tx, (long long)NT * DIM2,
              NT, DIM2, DIM, BSZ, stream);
  // 6) rep = LN(rep) in place (attention norm), then tx = concat(x1e,x2)+rep
  hipLaunchKernelGGL(layernorm_kernel, dim3(MROWS), dim3(256), 0, stream,
                     ws + off_tx, g_an, be_an, ws + off_tx, DIM2, 0);
  hipLaunchKernelGGL(tx_concat_add_kernel, dim3((MROWS * DIM2) / 256), dim3(256), 0, stream,
                     ws + off_x1e, x2, ws + off_tx);
  // 7) ntx = LN2(tx)   (spans Ksm+Qsm region, both dead now)
  hipLaunchKernelGGL(layernorm_kernel, dim3(MROWS), dim3(256), 0, stream,
                     ws + off_tx, g_n2, be_n2, ws + off_Ksm, DIM2, 0);
  // 8) h = ntx @ w_fc1 + b_fc1
  launch_gemm(ws + off_Ksm, 0, 0, DIM2, w_fc1, b_fc1, nullptr, ws + off_h, 0,
              MROWS, C2D, DIM2, 1, stream);
  // 9) dwph = depthwise_conv3d(h) + b_dw + h ; ax = gelu(LN(dwph))
  hipLaunchKernelGGL(dwconv_kernel, dim3((MROWS * C2D) / 256), dim3(256), 0, stream,
                     ws + off_h, w_dw, b_dw, ws + off_dwph);
  hipLaunchKernelGGL(layernorm_kernel, dim3(MROWS), dim3(256), 0, stream,
                     ws + off_dwph, g_m1, be_m1, ws + off_ax, C2D, 1);
  // 10) mx = tx + ax @ w_fc2 + b_fc2   (mx reuses ntx region)
  launch_gemm(ws + off_ax, 0, 0, C2D, w_fc2, b_fc2, ws + off_tx, ws + off_Ksm, 0,
              MROWS, DIM2, C2D, 1, stream);
  // 11) y = mx @ w_cat + b_cat        (y reuses n2 region)
  launch_gemm(ws + off_Ksm, 0, 0, DIM2, w_cat, b_cat, nullptr, ws + off_n2, 0,
              MROWS, DIM, DIM2, 1, stream);
  // 12) ye = y @ w_exp (no bias)      (ye reuses dwph region)
  launch_gemm(ws + off_n2, 0, 0, DIM, w_exp, nullptr, nullptr, ws + off_dwph, 0,
              MROWS, 4 * DIM, DIM, 1, stream);
  // 13) out = LN(ye viewed as [B, 8N, 128])
  hipLaunchKernelGGL(layernorm_kernel, dim3(BSZ * 8 * NT), dim3(256), 0, stream,
                     ws + off_dwph, g_up, be_up, (float*)d_out, DIM / 2, 0);
}
